// TransformerBlock_55954833932498
// MI455X (gfx1250) — compile-verified, matrix operand order checked by reading, not run on hardware
//
#include <hip/hip_runtime.h>

// ---------------------------------------------------------------------------
// Problem constants (from reference)
// ---------------------------------------------------------------------------
#define BATCH 2
#define SEQ   2048
#define DIM   1024
#define NH    16
#define HD    64
#define FFN   4096
#define ROWS  (BATCH * SEQ)   // 4096 tokens

// ---------------------------------------------------------------------------
// CDNA5 WMMA types
// ---------------------------------------------------------------------------
typedef __attribute__((ext_vector_type(16))) __bf16        v16bf;
typedef __attribute__((ext_vector_type(8)))  float         v8f;
typedef __attribute__((ext_vector_type(8)))  unsigned short v8us;

static __device__ __forceinline__ v8f wmma_bf16(v16bf a, v16bf b, v8f c) {
  // D = A(16x32 bf16) * B(32x16 bf16) + C(16x16 f32)
  return __builtin_amdgcn_wmma_f32_16x16x32_bf16(
      false, a, false, b, (short)0, c, false, false);
}

// float -> bf16 (round-to-nearest-even), stored as raw u16
static __device__ __forceinline__ unsigned short f2bf(float f) {
  unsigned int u = __float_as_uint(f);
  u += 0x7FFFu + ((u >> 16) & 1u);
  return (unsigned short)(u >> 16);
}

static __device__ __forceinline__ float gelu_exact(float v) {
  return 0.5f * v * (1.0f + erff(v * 0.70710678118654752f));
}

// Async global->LDS copy, 16B per lane.  Tracked by ASYNCcnt.
// (CDNA5 ISA op 98 GLOBAL_LOAD_ASYNC_TO_LDS_B128; inline asm keeps
//  it portable across the two toolchains' differing builtin surfaces.)
static __device__ __forceinline__ void async_copy_b128(
    unsigned short* lds_dst, const unsigned short* gsrc) {
  unsigned            lds = (unsigned)(size_t)lds_dst;       // low 32b = LDS offset
  unsigned long long  ga  = (unsigned long long)(size_t)gsrc;
  asm volatile("global_load_async_to_lds_b128 %0, %1, off"
               :: "v"(lds), "v"(ga) : "memory");
}
static __device__ __forceinline__ void wait_async0() {
  asm volatile("s_wait_asynccnt 0" ::: "memory");
}

// ---------------------------------------------------------------------------
// Weight convert + transpose: f32 [K,N] -> bf16 [N,K]
// ---------------------------------------------------------------------------
__global__ __launch_bounds__(256)
void wcvt_t_kernel(const float* __restrict__ in, unsigned short* __restrict__ out,
                   int K, int N) {
  size_t idx = (size_t)blockIdx.x * 256 + threadIdx.x;
  size_t total = (size_t)K * N;
  if (idx >= total) return;
  int k = (int)(idx / N);
  int n = (int)(idx % N);
  out[(size_t)n * K + k] = f2bf(in[idx]);
}

// ---------------------------------------------------------------------------
// LayerNorm (f32 in) -> bf16 out.  One 256-thread block (8 waves) per row.
// ---------------------------------------------------------------------------
__global__ __launch_bounds__(256)
void ln_bf16_kernel(const float* __restrict__ x, const float* __restrict__ g,
                    const float* __restrict__ bv, unsigned short* __restrict__ o) {
  __shared__ float sm[8];
  __shared__ float stats[2];
  const int row = blockIdx.x;
  const int tid = threadIdx.x;

  const float4 v = ((const float4*)(x + (size_t)row * DIM))[tid];
  float s = v.x + v.y + v.z + v.w;
  #pragma unroll
  for (int off = 16; off > 0; off >>= 1) s += __shfl_xor(s, off, 32);
  if ((tid & 31) == 0) sm[tid >> 5] = s;
  __syncthreads();
  if (tid == 0) {
    float t = 0.f;
    #pragma unroll
    for (int i = 0; i < 8; ++i) t += sm[i];
    stats[0] = t * (1.0f / DIM);
  }
  __syncthreads();
  const float mu = stats[0];

  float d0 = v.x - mu, d1 = v.y - mu, d2 = v.z - mu, d3 = v.w - mu;
  float q = d0 * d0 + d1 * d1 + d2 * d2 + d3 * d3;
  #pragma unroll
  for (int off = 16; off > 0; off >>= 1) q += __shfl_xor(q, off, 32);
  if ((tid & 31) == 0) sm[tid >> 5] = q;
  __syncthreads();
  if (tid == 0) {
    float t = 0.f;
    #pragma unroll
    for (int i = 0; i < 8; ++i) t += sm[i];
    stats[1] = rsqrtf(t * (1.0f / DIM) + 1e-5f);
  }
  __syncthreads();
  const float rs = stats[1];

  const int c = tid * 4;
  const float4 gv = ((const float4*)g)[tid];
  const float4 bb = ((const float4*)bv)[tid];
  const size_t base = (size_t)row * DIM + c;
  o[base + 0] = f2bf(d0 * rs * gv.x + bb.x);
  o[base + 1] = f2bf(d1 * rs * gv.y + bb.y);
  o[base + 2] = f2bf(d2 * rs * gv.z + bb.z);
  o[base + 3] = f2bf(d3 * rs * gv.w + bb.w);
}

// ---------------------------------------------------------------------------
// WMMA GEMM:  C[M,N] = A_bf16[M,K] @ Wt_bf16[N,K]^T + bias (+GELU) (+resid)
// Block tile 64x128, 256 threads = 8 waves (2 M-waves x 4 N-waves).
// Each wave owns 32x32 via four 16x16 f32 accumulators (2x2 fragment reuse).
// A tile (64x32) double-buffered in LDS, filled with async global->LDS copies
// overlapped with WMMA compute; ASYNCcnt fenced before each barrier.
// ---------------------------------------------------------------------------
template <bool GELU, bool RES, bool OUTBF>
__global__ __launch_bounds__(256)
void gemm_wmma_kernel(const unsigned short* __restrict__ A,
                      const unsigned short* __restrict__ Wt,
                      const float* __restrict__ bias,
                      const float* __restrict__ resid,
                      float* __restrict__ Cf,
                      unsigned short* __restrict__ Cb,
                      int M, int N, int K) {
  __shared__ __align__(64) unsigned short As[2][64 * 32];   // 2 x 4 KB

  const int tid  = threadIdx.x;
  const int lane = tid & 31;
  const int wave = tid >> 5;
  const int wm   = wave & 1;          // 0..1  (M sub-tile, 32 rows each)
  const int wn   = wave >> 1;         // 0..3  (N sub-tile, 32 cols each)
  const int l16  = lane & 15;
  const int lh   = lane >> 4;

  const long m0 = (long)blockIdx.y * 64;
  const long n0 = (long)blockIdx.x * 128;

  v8f acc[2][2];
  acc[0][0] = v8f{}; acc[0][1] = v8f{};
  acc[1][0] = v8f{}; acc[1][1] = v8f{};

  const int li = tid * 8;             // 2048 bf16 elements, 8 per thread
  const int ar = li >> 5;             // tile row 0..63
  const int ac = li & 31;             // tile col, step 8
  const unsigned short* agp = &A[(m0 + ar) * (long)K + ac];

  // Prologue: async-fill tile 0
  async_copy_b128(&As[0][li], agp);
  wait_async0();
  __syncthreads();

  int buf = 0;
  for (int k0 = 0; k0 < K; k0 += 32, buf ^= 1) {
    const bool more = (k0 + 32) < K;
    if (more) {
      // Fill the other buffer while this one is consumed
      async_copy_b128(&As[buf ^ 1][li], agp + k0 + 32);
      __builtin_prefetch(agp + k0 + 64, 0, 1);     // global_prefetch_b8
    }

    // A fragments (2 per wave) from LDS
    v16bf af0 = *(const v16bf*)&As[buf][(wm * 32 + l16) * 32 + lh * 16];
    v16bf af1 = *(const v16bf*)&As[buf][(wm * 32 + 16 + l16) * 32 + lh * 16];

    // B fragments (2 per wave) from L2-resident transposed weights
    const unsigned short* bp =
        &Wt[(n0 + wn * 32 + l16) * (long)K + k0 + lh * 16];
    v16bf bf0 = *(const v16bf*)bp;
    v16bf bf1 = *(const v16bf*)(bp + 16L * K);

    acc[0][0] = wmma_bf16(af0, bf0, acc[0][0]);
    acc[0][1] = wmma_bf16(af0, bf1, acc[0][1]);
    acc[1][0] = wmma_bf16(af1, bf0, acc[1][0]);
    acc[1][1] = wmma_bf16(af1, bf1, acc[1][1]);

    if (more) wait_async0();
    __syncthreads();
  }

  // Epilogue: C layout — lane half selects M offset 0/8, l16 selects N
  #pragma unroll
  for (int mi = 0; mi < 2; ++mi) {
    #pragma unroll
    for (int ni = 0; ni < 2; ++ni) {
      const long ncol = n0 + wn * 32 + ni * 16 + l16;
      const float bs = bias[ncol];
      #pragma unroll
      for (int r = 0; r < 8; ++r) {
        const long m = m0 + wm * 32 + mi * 16 + r + lh * 8;
        float v = acc[mi][ni][r] + bs;
        if (GELU) v = gelu_exact(v);
        if (RES)  v += resid[m * N + ncol];
        if (OUTBF) Cb[m * N + ncol] = f2bf(v);
        else       Cf[m * N + ncol] = v;
      }
    }
  }
}

// ---------------------------------------------------------------------------
// RoPE + head split.  qkv f32 [ROWS, 3*DIM] ->
//   Q,K bf16 [B,H,S,HD]  and  V bf16 transposed [B,H,HD,S]
// ---------------------------------------------------------------------------
__global__ __launch_bounds__(256)
void rope_split_kernel(const float* __restrict__ qkv,
                       unsigned short* __restrict__ Qb,
                       unsigned short* __restrict__ Kb,
                       unsigned short* __restrict__ Vt) {
  const int idx = blockIdx.x * 256 + threadIdx.x;   // B*S*H*HD = 2^22 total
  const int d = idx & (HD - 1);
  const int h = (idx >> 6) & (NH - 1);
  const int s = (idx >> 10) & (SEQ - 1);
  const int b = idx >> 21;

  const size_t base = ((size_t)(b * SEQ + s)) * (3 * DIM) + h * HD + d;
  const float q = qkv[base];
  const float k = qkv[base + DIM];
  const float v = qkv[base + 2 * DIM];

  const int   di  = d & 31;
  const size_t pb = (d & 32) ? (base - 32) : (base + 32);
  const float qp  = qkv[pb];
  const float kp  = qkv[pb + DIM];
  const float fr  = (float)s * __powf(10000.0f, -(float)di * (1.0f / 32.0f));
  float sn, cs;
  __sincosf(fr, &sn, &cs);
  const float qo = (d & 32) ? (q * cs + qp * sn) : (q * cs - qp * sn);
  const float ko = (d & 32) ? (k * cs + kp * sn) : (k * cs - kp * sn);

  const size_t bh = (size_t)b * NH + h;
  Qb[(bh * SEQ + s) * HD + d] = f2bf(qo);
  Kb[(bh * SEQ + s) * HD + d] = f2bf(ko);
  Vt[(bh * HD + d) * SEQ + s] = f2bf(v);
}

// ---------------------------------------------------------------------------
// Flash attention, one wave (32 lanes) per (b,h, 16-query tile).
// QK^T and P*V through v_wmma_f32_16x16x32_bf16; online softmax with
// width-16 shuffle row-reductions (wave32 C-layout halves).
// ---------------------------------------------------------------------------
__global__ __launch_bounds__(32)
void flash_attn_kernel(const unsigned short* __restrict__ Q,
                       const unsigned short* __restrict__ Kmat,
                       const unsigned short* __restrict__ Vt,
                       const unsigned char* __restrict__ mask,
                       unsigned short* __restrict__ ctx) {
  __shared__ __align__(64) unsigned short probs[16 * 32];   // 1 KB

  const int lane = threadIdx.x;
  const int l16  = lane & 15;
  const int lh   = lane >> 4;
  const int q0   = blockIdx.x * 16;
  const int bh   = blockIdx.y;          // b*NH + h
  const int b    = bh >> 4;
  const int h    = bh & (NH - 1);
  const float scale = 0.125f;           // 1/sqrt(64)

  // Q fragments (resident for the whole row of tiles): 16x64 -> two 16x32
  const unsigned short* qrow = Q + ((size_t)bh * SEQ + q0 + l16) * HD;
  const v16bf qf0 = *(const v16bf*)(qrow + lh * 16);
  const v16bf qf1 = *(const v16bf*)(qrow + 32 + lh * 16);

  v8f ctxa[4];
  #pragma unroll
  for (int t = 0; t < 4; ++t) ctxa[t] = v8f{};
  float mrow[8], lrow[8];
  #pragma unroll
  for (int r = 0; r < 8; ++r) { mrow[r] = -1e30f; lrow[r] = 0.0f; }

  const unsigned char* mp = mask + (size_t)b * SEQ;

  for (int kb = 0; kb < SEQ; kb += 32) {
    // ---- scores: 16 q x 32 keys = two 16x16 tiles, K-dim = HD = 64 ----
    v8f s0 = v8f{}, s1 = v8f{};
    const unsigned short* kp0 =
        Kmat + ((size_t)bh * SEQ + kb + l16) * HD + lh * 16;
    const unsigned short* kp1 = kp0 + 16 * HD;
    s0 = wmma_bf16(qf0, *(const v16bf*)kp0,        s0);
    s0 = wmma_bf16(qf1, *(const v16bf*)(kp0 + 32), s0);
    s1 = wmma_bf16(qf0, *(const v16bf*)kp1,        s1);
    s1 = wmma_bf16(qf1, *(const v16bf*)(kp1 + 32), s1);

    const float neg0 = mp[kb + l16]      ? -1e30f : 0.0f;
    const float neg1 = mp[kb + 16 + l16] ? -1e30f : 0.0f;

    // ---- online softmax, per output row (VGPR index r) ----
    #pragma unroll
    for (int r = 0; r < 8; ++r) {
      float a = s0[r] * scale + neg0;
      float c = s1[r] * scale + neg1;
      float mx = fmaxf(a, c);
      #pragma unroll
      for (int off = 8; off > 0; off >>= 1)
        mx = fmaxf(mx, __shfl_xor(mx, off, 16));
      const float mn    = fmaxf(mrow[r], mx);
      const float alpha = __expf(mrow[r] - mn);
      const float p0    = __expf(a - mn);
      const float p1    = __expf(c - mn);
      float sum = p0 + p1;
      #pragma unroll
      for (int off = 8; off > 0; off >>= 1)
        sum += __shfl_xor(sum, off, 16);
      lrow[r] = lrow[r] * alpha + sum;
      mrow[r] = mn;
      #pragma unroll
      for (int t = 0; t < 4; ++t) ctxa[t][r] *= alpha;

      const int row = r + lh * 8;
      probs[row * 32 + l16]      = f2bf(p0);
      probs[row * 32 + 16 + l16] = f2bf(p1);
    }
    __syncthreads();

    // ---- P (16x32) @ V (32 x 64): four 16x16 ctx tiles ----
    const v16bf pf = *(const v16bf*)&probs[l16 * 32 + lh * 16];
    #pragma unroll
    for (int t = 0; t < 4; ++t) {
      const unsigned short* vp =
          Vt + ((size_t)bh * HD + t * 16 + l16) * SEQ + kb + lh * 16;
      ctxa[t] = wmma_bf16(pf, *(const v16bf*)vp, ctxa[t]);
    }
    __syncthreads();
  }

  // ---- normalize and emit ctx as bf16 [B*S, DIM] (= [b,s,h,hd]) ----
  #pragma unroll
  for (int r = 0; r < 8; ++r) {
    const float inv = 1.0f / lrow[r];
    const size_t base =
        ((size_t)b * SEQ + q0 + r + lh * 8) * DIM + h * HD;
    #pragma unroll
    for (int t = 0; t < 4; ++t)
      ctx[base + t * 16 + l16] = f2bf(ctxa[t][r] * inv);
  }
}

// ---------------------------------------------------------------------------
// Host-side orchestration
// ---------------------------------------------------------------------------
extern "C" void kernel_launch(void* const* d_in, const int* in_sizes, int n_in,
                              void* d_out, int out_size, void* d_ws, size_t ws_size,
                              hipStream_t stream) {
  (void)in_sizes; (void)n_in; (void)out_size; (void)ws_size;

  const float*         x     = (const float*)d_in[0];
  const unsigned char* pmask = (const unsigned char*)d_in[1];
  const float* qkv_w = (const float*)d_in[2];
  const float* qkv_b = (const float*)d_in[3];
  const float* out_w = (const float*)d_in[4];
  const float* out_b = (const float*)d_in[5];
  const float* ln1_g = (const float*)d_in[6];
  const float* ln1_b = (const float*)d_in[7];
  const float* ln2_g = (const float*)d_in[8];
  const float* ln2_b = (const float*)d_in[9];
  const float* w1    = (const float*)d_in[10];
  const float* b1    = (const float*)d_in[11];
  const float* w2    = (const float*)d_in[12];
  const float* b2    = (const float*)d_in[13];
  float* out = (float*)d_out;

  // Workspace carve-up (aligned 256B)
  char* ws = (char*)d_ws;
  auto carve = [&](size_t bytes) {
    char* p = ws;
    ws += (bytes + 255) & ~(size_t)255;
    return p;
  };
  unsigned short* h1    = (unsigned short*)carve((size_t)ROWS * DIM * 2);
  unsigned short* qkvwt = (unsigned short*)carve((size_t)DIM * 3 * DIM * 2);
  unsigned short* outwt = (unsigned short*)carve((size_t)DIM * DIM * 2);
  unsigned short* w1t   = (unsigned short*)carve((size_t)DIM * FFN * 2);
  unsigned short* w2t   = (unsigned short*)carve((size_t)FFN * DIM * 2);
  float*          qkvf  = (float*)carve((size_t)ROWS * 3 * DIM * 4); // reused as h3
  unsigned short* Qb    = (unsigned short*)carve((size_t)ROWS * DIM * 2);
  unsigned short* Kb    = (unsigned short*)carve((size_t)ROWS * DIM * 2);
  unsigned short* Vt    = (unsigned short*)carve((size_t)ROWS * DIM * 2);
  unsigned short* ctxb  = (unsigned short*)carve((size_t)ROWS * DIM * 2);
  float*          x1    = (float*)carve((size_t)ROWS * DIM * 4);
  unsigned short* h2    = (unsigned short*)carve((size_t)ROWS * DIM * 2);
  unsigned short* h3    = (unsigned short*)qkvf;   // overlay: qkvf dead after RoPE

  // 1) weights -> bf16 transposed [N,K]
  wcvt_t_kernel<<<(DIM * 3 * DIM) / 256, 256, 0, stream>>>(qkv_w, qkvwt, DIM, 3 * DIM);
  wcvt_t_kernel<<<(DIM * DIM) / 256,     256, 0, stream>>>(out_w, outwt, DIM, DIM);
  wcvt_t_kernel<<<(DIM * FFN) / 256,     256, 0, stream>>>(w1,    w1t,   DIM, FFN);
  wcvt_t_kernel<<<(FFN * DIM) / 256,     256, 0, stream>>>(w2,    w2t,   FFN, DIM);

  // 2) LN1 -> bf16
  ln_bf16_kernel<<<ROWS, 256, 0, stream>>>(x, ln1_g, ln1_b, h1);

  // 3) QKV projection: [4096,1024] @ [1024,3072]
  gemm_wmma_kernel<false, false, false>
      <<<dim3((3 * DIM) / 128, ROWS / 64), 256, 0, stream>>>(
          h1, qkvwt, qkv_b, nullptr, qkvf, nullptr, ROWS, 3 * DIM, DIM);

  // 4) RoPE + split heads
  rope_split_kernel<<<(ROWS * DIM) / 256, 256, 0, stream>>>(qkvf, Qb, Kb, Vt);

  // 5) attention
  flash_attn_kernel<<<dim3(SEQ / 16, BATCH * NH), 32, 0, stream>>>(
      Qb, Kb, Vt, pmask, ctxb);

  // 6) out-proj + residual(x) -> x1
  gemm_wmma_kernel<false, true, false>
      <<<dim3(DIM / 128, ROWS / 64), 256, 0, stream>>>(
          ctxb, outwt, out_b, x, x1, nullptr, ROWS, DIM, DIM);

  // 7) LN2 -> bf16
  ln_bf16_kernel<<<ROWS, 256, 0, stream>>>(x1, ln2_g, ln2_b, h2);

  // 8) FFN1 + exact GELU -> bf16
  gemm_wmma_kernel<true, false, true>
      <<<dim3(FFN / 128, ROWS / 64), 256, 0, stream>>>(
          h2, w1t, b1, nullptr, nullptr, h3, ROWS, FFN, DIM);

  // 9) FFN2 + residual(x1) -> out
  gemm_wmma_kernel<false, true, false>
      <<<dim3(DIM / 128, ROWS / 64), 256, 0, stream>>>(
          h3, w2t, b2, x1, out, nullptr, ROWS, DIM, FFN);
}
